// GQAAttention_81114752352301
// MI455X (gfx1250) — matrix operand order repaired and checked by reading, not compiled, not run on hardware
//
#include <hip/hip_runtime.h>
#include <hip/hip_bf16.h>
#include <cstdint>
#include <cstddef>

// ---------------------------------------------------------------------------
// CDNA5 (gfx1250) GQA attention layer:
//   bf16 convert/transpose -> QKV WMMA GEMMs (async-LDS double buffered)
//   -> RoPE -> flash attention (async-LDS K/V staging, online softmax)
//   -> output projection WMMA GEMM.
// Matrix math: v_wmma_f32_16x16x32_bf16 (wave32).
// Data movement: global_load_async_to_lds_b128 + s_wait_asynccnt (ASYNCcnt).
// ---------------------------------------------------------------------------

typedef __attribute__((ext_vector_type(16))) __bf16 v16bf;
typedef __attribute__((ext_vector_type(8)))  float  v8f;

constexpr int DIM_ = 2048;
constexpr int NH_  = 16;
constexpr int NKV_ = 4;
constexpr int HD_  = 128;
constexpr int NREP_ = NH_ / NKV_;
constexpr int B_   = 2;
constexpr int T_   = 2048;

__device__ __forceinline__ unsigned short f2bf(float f) {
  unsigned int u = __float_as_uint(f);
  u += 0x7FFFu + ((u >> 16) & 1u);          // round-to-nearest-even
  return (unsigned short)(u >> 16);
}

union FragU { uint4 u[2]; v16bf v; };

__device__ __forceinline__ v16bf ldfrag(const unsigned short* p0,
                                        const unsigned short* p1) {
  FragU f;
  f.u[0] = *(const uint4*)p0;
  f.u[1] = *(const uint4*)p1;
  return f.v;
}

// --- CDNA5 async global->LDS copy (ASYNCcnt) -------------------------------
// VDST VGPR holds the LDS byte offset (low 32 bits of the flat shared addr),
// VADDR holds the 64-bit global address.  §15.18.3 op 98.
__device__ __forceinline__ void async_cp128(void* lds, const void* g) {
  unsigned loff = (unsigned)(size_t)lds;
  asm volatile("global_load_async_to_lds_b128 %0, %1, off"
               :: "v"(loff), "v"(g) : "memory");
}
__device__ __forceinline__ void wait_async_le8() {
  asm volatile("s_wait_asynccnt 8" ::: "memory");
}
__device__ __forceinline__ void wait_async_0() {
  asm volatile("s_wait_asynccnt 0" ::: "memory");
}

// ---------------------------------------------------------------------------
// f32 -> bf16 elementwise (vectorized), for x.
// ---------------------------------------------------------------------------
__global__ void cvt_bf16_kernel(const float* __restrict__ src,
                                unsigned short* __restrict__ dst) {
  const int i = blockIdx.x * blockDim.x + threadIdx.x;  // float4 index
  const float4 v = ((const float4*)src)[i];
  uint2 p;
  p.x = (unsigned)f2bf(v.x) | ((unsigned)f2bf(v.y) << 16);
  p.y = (unsigned)f2bf(v.z) | ((unsigned)f2bf(v.w) << 16);
  ((uint2*)dst)[i] = p;
}

// ---------------------------------------------------------------------------
// Weights: f32 [K][N] -> bf16 transposed [N][K]  (coalesced via LDS tile).
// ---------------------------------------------------------------------------
__global__ __launch_bounds__(256)
void cvt_transpose_kernel(const float* __restrict__ src,
                          unsigned short* __restrict__ dst, int K, int N) {
  __shared__ unsigned short tile[32][33];
  const int kt = blockIdx.y * 32, nt = blockIdx.x * 32;
  const int tx = threadIdx.x, ty = threadIdx.y;   // (32, 8)
  #pragma unroll
  for (int i = 0; i < 32; i += 8)
    tile[ty + i][tx] = f2bf(src[(size_t)(kt + ty + i) * N + nt + tx]);
  __syncthreads();
  #pragma unroll
  for (int i = 0; i < 32; i += 8)
    dst[(size_t)(nt + ty + i) * K + kt + tx] = tile[tx][ty + i];
}

// ---------------------------------------------------------------------------
// GEMM: C[M][N] = A[M][K] * BT[N][K]^T   (A,BT bf16; C f32)
// 128 thr = 4 waves; block tile 64x64; wave tile 32x32; k-chunk 64.
// Double-buffered async stage-in: 8 x b128 per thread per tile pair.
// ---------------------------------------------------------------------------
__global__ __launch_bounds__(128)
void gemm_bf_wmma(const unsigned short* __restrict__ A,
                  const unsigned short* __restrict__ BT,
                  float* __restrict__ C, int M, int N, int K) {
  __shared__ __align__(16) unsigned short As[2][64][72];  // [m][k]
  __shared__ __align__(16) unsigned short Bs[2][64][72];  // [n][k]

  const int tid  = threadIdx.x;
  const int lane = tid & 31;
  const int wave = tid >> 5;
  const int half = lane >> 4;
  const int l16  = lane & 15;
  const int wm = wave >> 1, wn = wave & 1;
  const int m0 = blockIdx.y * 64;
  const int n0 = blockIdx.x * 64;

  v8f acc[2][2] = {};

  auto stage = [&](int k0, int buf) {
    #pragma unroll
    for (int i = 0; i < 4; ++i) {
      const int q = tid + 128 * i;          // 0..511 : 16B chunks of 8 KiB
      const int row = q >> 3, c8 = q & 7;   // 8 chunks per 64-elem row
      async_cp128(&As[buf][row][c8 * 8],
                  A + (size_t)(m0 + row) * K + k0 + c8 * 8);
    }
    #pragma unroll
    for (int i = 0; i < 4; ++i) {
      const int q = tid + 128 * i;
      const int row = q >> 3, c8 = q & 7;
      async_cp128(&Bs[buf][row][c8 * 8],
                  BT + (size_t)(n0 + row) * K + k0 + c8 * 8);
    }
  };

  stage(0, 0);
  const int niter = K / 64;
  for (int it = 0; it < niter; ++it) {
    const int buf = it & 1;
    if (it + 1 < niter) { stage((it + 1) * 64, buf ^ 1); wait_async_le8(); }
    else                { wait_async_0(); }
    __syncthreads();

    #pragma unroll
    for (int kc = 0; kc < 2; ++kc) {
      v16bf af[2], bfr[2];
      #pragma unroll
      for (int mt = 0; mt < 2; ++mt) {
        const unsigned short* r = &As[buf][32 * wm + 16 * mt + l16][32 * kc];
        af[mt] = ldfrag(r + half * 8, r + 16 + half * 8);
      }
      #pragma unroll
      for (int nt = 0; nt < 2; ++nt) {
        const unsigned short* r = &Bs[buf][32 * wn + 16 * nt + l16][32 * kc];
        bfr[nt] = ldfrag(r + half * 8, r + 16 + half * 8);
      }
      #pragma unroll
      for (int mt = 0; mt < 2; ++mt)
        #pragma unroll
        for (int nt = 0; nt < 2; ++nt)
          acc[mt][nt] = __builtin_amdgcn_wmma_f32_16x16x32_bf16(
              false, af[mt], false, bfr[nt], (short)0, acc[mt][nt], false, false);
    }
    __syncthreads();
  }

  #pragma unroll
  for (int mt = 0; mt < 2; ++mt)
    #pragma unroll
    for (int nt = 0; nt < 2; ++nt)
      #pragma unroll
      for (int r = 0; r < 8; ++r) {
        const int row = m0 + 32 * wm + 16 * mt + r + 8 * half;
        const int col = n0 + 32 * wn + 16 * nt + l16;
        C[(size_t)row * N + col] = acc[mt][nt][r];
      }
}

// ---------------------------------------------------------------------------
// RoPE Q:  qraw [B,T,NH,HD] f32 -> qbf [B,NH,T,HD] bf16 (rotated)
// ---------------------------------------------------------------------------
__global__ void rope_q_kernel(const float* __restrict__ qraw,
                              const float* __restrict__ fr,
                              unsigned short* __restrict__ qbf) {
  const int idx = blockIdx.x * blockDim.x + threadIdx.x; // pair index
  const int d2 = idx & 63;
  const int h  = (idx >> 6) & (NH_ - 1);
  const int t  = (idx >> 10) & (T_ - 1);
  const int b  = idx >> 21;
  const float c = fr[(t * 64 + d2) * 2 + 0];
  const float s = fr[(t * 64 + d2) * 2 + 1];
  const size_t src = (((size_t)(b * T_ + t) * NH_ + h) * HD_) + 2 * d2;
  const float x0 = qraw[src], x1 = qraw[src + 1];
  const size_t dst = (((size_t)(b * NH_ + h) * T_ + t) * HD_) + 2 * d2;
  qbf[dst]     = f2bf(x0 * c - x1 * s);
  qbf[dst + 1] = f2bf(x0 * s + x1 * c);
}

// ---------------------------------------------------------------------------
// RoPE K + layout V:
//   kraw/vraw [B,T,NKV,HD] f32
//   -> kcache/vcache [B,NKV,T,HD] f32 (module outputs)
//   -> kbf [B,NKV,T,HD] bf16, vbft [B,NKV,HD,T] bf16 (d-major for P@V frags)
// ---------------------------------------------------------------------------
__global__ void rope_kv_kernel(const float* __restrict__ kraw,
                               const float* __restrict__ vraw,
                               const float* __restrict__ fr,
                               unsigned short* __restrict__ kbf,
                               unsigned short* __restrict__ vbft,
                               float* __restrict__ kcache,
                               float* __restrict__ vcache) {
  const int idx = blockIdx.x * blockDim.x + threadIdx.x; // pair index
  const int d2 = idx & 63;
  const int g  = (idx >> 6) & (NKV_ - 1);
  const int t  = (idx >> 8) & (T_ - 1);
  const int b  = idx >> 19;
  const float c = fr[(t * 64 + d2) * 2 + 0];
  const float s = fr[(t * 64 + d2) * 2 + 1];
  const size_t src = (((size_t)(b * T_ + t) * NKV_ + g) * HD_) + 2 * d2;
  const size_t bg  = (size_t)(b * NKV_ + g);
  const size_t dstk = ((bg * T_ + t) * HD_) + 2 * d2;
  const float k0 = kraw[src], k1 = kraw[src + 1];
  const float r0 = k0 * c - k1 * s, r1 = k0 * s + k1 * c;
  kcache[dstk] = r0;        kcache[dstk + 1] = r1;
  kbf[dstk]    = f2bf(r0);  kbf[dstk + 1]    = f2bf(r1);
  const float v0 = vraw[src], v1 = vraw[src + 1];
  vcache[dstk] = v0;        vcache[dstk + 1] = v1;
  const size_t dv = (bg * HD_ + 2 * d2) * T_ + t;
  vbft[dv]      = f2bf(v0);
  vbft[dv + T_] = f2bf(v1);
}

// ---------------------------------------------------------------------------
// Flash attention, 4 waves / block, 64 q-rows per (b,h) block.
// K/V blocks (32 keys) staged to LDS with double-buffered async copies and
// shared by all 4 waves.  Online softmax; WMMA for S=Q.K^T and O+=P.V.
// ---------------------------------------------------------------------------
__global__ __launch_bounds__(128)
void flash_attn_kernel(const unsigned short* __restrict__ qbf,
                       const unsigned short* __restrict__ kbf,
                       const unsigned short* __restrict__ vbft,
                       unsigned short* __restrict__ attn) {
  __shared__ __align__(16) unsigned short Ks[2][32][136];  // [key][dim]+pad
  __shared__ __align__(16) unsigned short Vs[2][128][40];  // [dim][key]+pad
  __shared__ __align__(16) unsigned short Ps[4][16][40];   // per-wave P tile

  const int tid  = threadIdx.x;
  const int lane = tid & 31;
  const int wave = tid >> 5;
  const int half = lane >> 4;
  const int l16  = lane & 15;
  const int q0  = blockIdx.x * 64;
  const int q0w = q0 + 16 * wave;
  const int h   = blockIdx.y;
  const int b   = blockIdx.z;
  const int g   = h / NREP_;
  const size_t bh = (size_t)(b * NH_ + h);
  const size_t bg = (size_t)(b * NKV_ + g);

  // Q fragments for this wave's 16 rows (4 x 16x32 bf16 A-frags)
  v16bf qf[4];
  #pragma unroll
  for (int c = 0; c < 4; ++c) {
    const unsigned short* r = qbf + (bh * T_ + q0w + l16) * HD_ + 32 * c;
    qf[c] = ldfrag(r + half * 8, r + 16 + half * 8);
  }

  v8f o[8] = {};
  float mrow[8], lrow[8];
  #pragma unroll
  for (int r = 0; r < 8; ++r) { mrow[r] = -3.0e38f; lrow[r] = 0.0f; }

  auto stage = [&](int s0, int buf) {
    // K block: 32 keys x 128 dims bf16 = 8 KiB contiguous
    const char* kb = (const char*)(kbf + (bg * T_ + s0) * HD_);
    #pragma unroll
    for (int i = 0; i < 4; ++i) {
      const int q = tid + 128 * i;               // 0..511
      const int row = q >> 4, c16 = q & 15;      // 16 chunks / 256B row
      async_cp128(&Ks[buf][row][c16 * 8], kb + q * 16);
    }
    // V block: 128 dims x 32 keys bf16 (d-major rows of 64B)
    #pragma unroll
    for (int i = 0; i < 4; ++i) {
      const int q = tid + 128 * i;
      const int row = q >> 2, c4 = q & 3;        // 4 chunks / 64B row
      async_cp128(&Vs[buf][row][c4 * 8],
                  (const char*)(vbft + (bg * HD_ + row) * T_ + s0) + c4 * 16);
    }
  };

  const float scale = 0.08838834764831845f;      // 1/sqrt(128)
  const int nblk = (q0 >> 5) + 2;                // keys 0 .. q0+63

  stage(0, 0);
  for (int j = 0; j < nblk; ++j) {
    const int s0 = j * 32;
    const int buf = j & 1;
    if (j + 1 < nblk) { stage((j + 1) * 32, buf ^ 1); wait_async_le8(); }
    else              { wait_async_0(); }
    __syncthreads();

    if (s0 <= q0w + 15) {                        // causal: block has live keys
      // ---- S = Q * K^T (two 16x16 tiles) ----
      v8f st[2];
      #pragma unroll
      for (int nt = 0; nt < 2; ++nt) {
        v8f acc = {};
        const unsigned short* kr = &Ks[buf][16 * nt + l16][8 * half];
        #pragma unroll
        for (int c = 0; c < 4; ++c) {
          v16bf bfr = ldfrag(kr + 32 * c, kr + 32 * c + 16);
          acc = __builtin_amdgcn_wmma_f32_16x16x32_bf16(
              false, qf[c], false, bfr, (short)0, acc, false, false);
        }
        st[nt] = acc;
      }

      // ---- scale + causal mask + row max ----
      float rmax[8];
      #pragma unroll
      for (int r = 0; r < 8; ++r) {
        const int qr = q0w + r + 8 * half;
        #pragma unroll
        for (int nt = 0; nt < 2; ++nt) {
          const int sc = s0 + 16 * nt + l16;
          float v = st[nt][r] * scale;
          if (sc > qr) v = -1.0e30f;
          st[nt][r] = v;
        }
        rmax[r] = fmaxf(st[0][r], st[1][r]);
      }
      #pragma unroll
      for (int off = 8; off >= 1; off >>= 1)
        #pragma unroll
        for (int r = 0; r < 8; ++r)
          rmax[r] = fmaxf(rmax[r], __shfl_xor(rmax[r], off, 32));

      // ---- online softmax update ----
      float rsum[8];
      #pragma unroll
      for (int r = 0; r < 8; ++r) {
        const float mn = fmaxf(mrow[r], rmax[r]);
        const float alpha = __expf(mrow[r] - mn);
        mrow[r] = mn;
        lrow[r] *= alpha;
        #pragma unroll
        for (int n8 = 0; n8 < 8; ++n8) o[n8][r] *= alpha;
        const float p0 = __expf(st[0][r] - mn);
        const float p1 = __expf(st[1][r] - mn);
        st[0][r] = p0; st[1][r] = p1;
        rsum[r] = p0 + p1;
      }
      #pragma unroll
      for (int off = 8; off >= 1; off >>= 1)
        #pragma unroll
        for (int r = 0; r < 8; ++r)
          rsum[r] += __shfl_xor(rsum[r], off, 32);
      #pragma unroll
      for (int r = 0; r < 8; ++r) lrow[r] += rsum[r];

      // ---- P (C-layout f32) -> LDS -> A-fragment (bf16 16x32) ----
      #pragma unroll
      for (int nt = 0; nt < 2; ++nt)
        #pragma unroll
        for (int r = 0; r < 8; ++r)
          Ps[wave][r + 8 * half][16 * nt + l16] = f2bf(st[nt][r]);
      asm volatile("s_wait_dscnt 0" ::: "memory");
      const unsigned short* pr = &Ps[wave][l16][0];
      v16bf pf = ldfrag(pr + half * 8, pr + 16 + half * 8);

      // ---- O += P * V ----
      #pragma unroll
      for (int n8 = 0; n8 < 8; ++n8) {
        const unsigned short* vr = &Vs[buf][16 * n8 + l16][8 * half];
        v16bf bfv = ldfrag(vr, vr + 16);
        o[n8] = __builtin_amdgcn_wmma_f32_16x16x32_bf16(
            false, pf, false, bfv, (short)0, o[n8], false, false);
      }
    }
    __syncthreads();   // everyone done with buf before it is restaged
  }

  // ---- normalize + store bf16 [B,T,NH*HD] ----
  #pragma unroll
  for (int n8 = 0; n8 < 8; ++n8)
    #pragma unroll
    for (int r = 0; r < 8; ++r) {
      const int t = q0w + r + 8 * half;
      const float val = o[n8][r] / lrow[r];
      attn[(size_t)(b * T_ + t) * (NH_ * HD_) + h * HD_ + 16 * n8 + l16] =
          f2bf(val);
    }
}

// ---------------------------------------------------------------------------
extern "C" void kernel_launch(void* const* d_in, const int* in_sizes, int n_in,
                              void* d_out, int out_size, void* d_ws,
                              size_t ws_size, hipStream_t stream) {
  const float* x  = (const float*)d_in[0];
  const float* wq = (const float*)d_in[1];
  const float* wk = (const float*)d_in[2];
  const float* wv = (const float*)d_in[3];
  const float* wo = (const float*)d_in[4];
  const float* fr = (const float*)d_in[5];

  char* ws = (char*)d_ws;
  unsigned short* xbf  = (unsigned short*)(ws + 0);          // 16 MiB
  unsigned short* wqT  = (unsigned short*)(ws + 16777216);   //  8 MiB [2048][2048]
  unsigned short* wkT  = (unsigned short*)(ws + 25165824);   //  2 MiB [512][2048]
  unsigned short* wvT  = (unsigned short*)(ws + 27262976);   //  2 MiB
  unsigned short* woT  = (unsigned short*)(ws + 29360128);   //  8 MiB [2048][2048]
  float* qraw = (float*)(ws + 37748736);                     // 32 MiB
  float* kraw = (float*)(ws + 71303168);                     //  8 MiB
  float* vraw = (float*)(ws + 79691776);                     //  8 MiB
  unsigned short* qbf  = (unsigned short*)(ws + 88080384);   // 16 MiB
  unsigned short* kbf  = (unsigned short*)(ws + 104857600);  //  4 MiB
  unsigned short* vbft = (unsigned short*)(ws + 109051904);  //  4 MiB
  unsigned short* attn = (unsigned short*)(ws + 113246208);  // 16 MiB

  float* out    = (float*)d_out;                          // [B,T,DIM]
  float* kcache = out + (size_t)B_ * T_ * DIM_;           // [B,NKV,T,HD]
  float* vcache = kcache + (size_t)B_ * NKV_ * T_ * HD_;  // [B,NKV,T,HD]

  // 1) precision conversion + weight transposes (one-time, bandwidth-cheap)
  cvt_bf16_kernel<<<(B_ * T_ * DIM_ / 4) / 256, 256, 0, stream>>>(x, xbf);
  cvt_transpose_kernel<<<dim3(DIM_ / 32, DIM_ / 32), dim3(32, 8), 0, stream>>>(
      wq, wqT, DIM_, NH_ * HD_);
  cvt_transpose_kernel<<<dim3((NKV_ * HD_) / 32, DIM_ / 32), dim3(32, 8), 0, stream>>>(
      wk, wkT, DIM_, NKV_ * HD_);
  cvt_transpose_kernel<<<dim3((NKV_ * HD_) / 32, DIM_ / 32), dim3(32, 8), 0, stream>>>(
      wv, wvT, DIM_, NKV_ * HD_);
  cvt_transpose_kernel<<<dim3(DIM_ / 32, (NH_ * HD_) / 32), dim3(32, 8), 0, stream>>>(
      wo, woT, NH_ * HD_, DIM_);

  // 2) QKV projections (WMMA, async-LDS double buffered)
  const dim3 gblk(128);
  gemm_bf_wmma<<<dim3(DIM_ / 64, (B_ * T_) / 64), gblk, 0, stream>>>(
      xbf, wqT, qraw, B_ * T_, NH_ * HD_, DIM_);
  gemm_bf_wmma<<<dim3((NKV_ * HD_) / 64, (B_ * T_) / 64), gblk, 0, stream>>>(
      xbf, wkT, kraw, B_ * T_, NKV_ * HD_, DIM_);
  gemm_bf_wmma<<<dim3((NKV_ * HD_) / 64, (B_ * T_) / 64), gblk, 0, stream>>>(
      xbf, wvT, vraw, B_ * T_, NKV_ * HD_, DIM_);

  // 3) RoPE + cache emission + attention-friendly layouts
  rope_q_kernel<<<(B_ * T_ * NH_ * (HD_ / 2)) / 256, 256, 0, stream>>>(qraw, fr, qbf);
  rope_kv_kernel<<<(B_ * T_ * NKV_ * (HD_ / 2)) / 256, 256, 0, stream>>>(
      kraw, vraw, fr, kbf, vbft, kcache, vcache);

  // 4) Flash attention (4 waves / 64 q-rows per block, shared async K/V)
  flash_attn_kernel<<<dim3(T_ / 64, NH_, B_), 128, 0, stream>>>(qbf, kbf, vbft, attn);

  // 5) Output projection
  gemm_bf_wmma<<<dim3(DIM_ / 64, (B_ * T_) / 64), gblk, 0, stream>>>(
      attn, woT, out, B_ * T_, DIM_, NH_ * HD_);
}
